// GridSampleConv_67723044323339
// MI455X (gfx1250) — matrix operand order repaired and checked by reading, not compiled, not run on hardware
//
#include <hip/hip_runtime.h>
#include <stdint.h>

#define N_PTS 131072
#define M_Q   65536
#define NNB   32
#define KPN   15
#define IN_F  64
#define OUT_F 128
#define NEG_SLOPE 0.01f
// KERNEL_RADIUS = 0.0375 ; KP_EXTENT = R/(15^(1/3)-1)*1.5
#define KP_EXTENT_INV (1.0f/(0.0375f/(2.4662120743304703f-1.0f)*1.5f))

typedef __attribute__((ext_vector_type(16))) __bf16 v16bf;
typedef __attribute__((ext_vector_type(2)))  __bf16 v2bf;
typedef __attribute__((ext_vector_type(8)))  float  v8f;

union BFV { v16bf v; unsigned int u[8]; uint4 q[2]; };

// Native f32 -> bf16 conversions (lower to v_cvt_*bf16_f32 on gfx1250).
__device__ __forceinline__ unsigned short f2bf(float f){
  __bf16 b = (__bf16)f;
  return __builtin_bit_cast(unsigned short, b);
}
__device__ __forceinline__ unsigned int pack2bf(float a, float b){
  v2bf t; t.x = (__bf16)a; t.y = (__bf16)b;
  return __builtin_bit_cast(unsigned int, t);
}
// Branch-free LeakyReLU: max(x,0) + slope*min(x,0)
__device__ __forceinline__ float leaky(float x){
  return __builtin_fmaf(NEG_SLOPE, fminf(x, 0.f), fmaxf(x, 0.f));
}

// Build a WMMA-A (16x16x32 bf16) operand for one lane from two contiguous
// 8-float chunks of a row: halves[0..7]=p1[0..7], halves[8..15]=p2[0..7].
__device__ __forceinline__ void fill_a_f32(BFV& a, const float* p1, const float* p2){
  float4 x0 = *(const float4*)p1, x1 = *(const float4*)(p1 + 4);
  float4 y0 = *(const float4*)p2, y1 = *(const float4*)(p2 + 4);
  a.u[0] = pack2bf(x0.x, x0.y); a.u[1] = pack2bf(x0.z, x0.w);
  a.u[2] = pack2bf(x1.x, x1.y); a.u[3] = pack2bf(x1.z, x1.w);
  a.u[4] = pack2bf(y0.x, y0.y); a.u[5] = pack2bf(y0.z, y0.w);
  a.u[6] = pack2bf(y1.x, y1.y); a.u[7] = pack2bf(y1.z, y1.w);
}

// ---------------------------------------------------------------------------
// Workspace layout (bytes)
// ---------------------------------------------------------------------------
static const size_t SZ_FEATS    = (size_t)(N_PTS + 1) * OUT_F * 2;   // bf16 feats + shadow row
static const size_t OFF_PKW     = SZ_FEATS;                          // 64 inner-tiles
static const size_t SZ_PKW      = (size_t)64 * 8 * 32 * 8 * 4;
static const size_t OFF_PKWPOST = OFF_PKW + SZ_PKW;
static const size_t SZ_PKWPOST  = (size_t)4 * 8 * 32 * 8 * 4;
static const size_t OFF_PKWPRE  = OFF_PKWPOST + SZ_PKWPOST;
static const size_t SZ_PKWPRE   = (size_t)2 * 8 * 32 * 8 * 4;
static const size_t OFF_PKWSC   = OFF_PKWPRE + SZ_PKWPRE;
static const size_t SZ_PKWSC    = (size_t)2 * 8 * 32 * 8 * 4;

// ---------------------------------------------------------------------------
// Prep: repack all weights into bf16 WMMA-B lane order; zero the shadow row.
// B-tile dword index: ((t*8 + ot)*32 + lane)*8 + v
// holds elements (inner = t*32 + g*8 + 2v + 8*(v>>2) + {0,1}, col = ot*16 + n)
// ---------------------------------------------------------------------------
__global__ void prep_pack_kernel(const float* __restrict__ W_pre,
                                 const float* __restrict__ kp_w,
                                 const float* __restrict__ W_post,
                                 const float* __restrict__ W_sc,
                                 unsigned short* __restrict__ feats_bf,
                                 unsigned int* __restrict__ pkW,
                                 unsigned int* __restrict__ pkWpost,
                                 unsigned int* __restrict__ pkWpre,
                                 unsigned int* __restrict__ pkWsc)
{
  int i = blockIdx.x * blockDim.x + threadIdx.x;
  if (i < 64 * 2048) {                       // kp_weights, inner index = c*16 + k
    int v = i & 7, lane = (i >> 3) & 31, ot = (i >> 8) & 7, t = i >> 11;
    int g = lane >> 4, n = lane & 15;
    int c = 2 * t + (v >> 2), o = ot * 16 + n;
    int k0 = 8 * g + 2 * v - 8 * (v >> 2), k1 = k0 + 1;
    float a = (k0 < KPN) ? kp_w[((size_t)k0 * OUT_F + c) * OUT_F + o] : 0.f;
    float b = (k1 < KPN) ? kp_w[((size_t)k1 * OUT_F + c) * OUT_F + o] : 0.f;
    pkW[i] = pack2bf(a, b);
    return;
  }
  i -= 64 * 2048;
  if (i < 4 * 2048) {                        // W_post [128][128]
    int v = i & 7, lane = (i >> 3) & 31, ot = (i >> 8) & 7, t = i >> 11;
    int g = lane >> 4, n = lane & 15;
    int c0 = t * 32 + g * 8 + 2 * v + 8 * (v >> 2), o = ot * 16 + n;
    pkWpost[i] = pack2bf(W_post[c0 * OUT_F + o], W_post[(c0 + 1) * OUT_F + o]);
    return;
  }
  i -= 4 * 2048;
  if (i < 2 * 2048) {                        // W_pre [64][128]
    int v = i & 7, lane = (i >> 3) & 31, ot = (i >> 8) & 7, t = i >> 11;
    int g = lane >> 4, n = lane & 15;
    int c0 = t * 32 + g * 8 + 2 * v + 8 * (v >> 2), o = ot * 16 + n;
    pkWpre[i] = pack2bf(W_pre[c0 * OUT_F + o], W_pre[(c0 + 1) * OUT_F + o]);
    return;
  }
  i -= 2 * 2048;
  if (i < 2 * 2048) {                        // W_sc [64][128]
    int v = i & 7, lane = (i >> 3) & 31, ot = (i >> 8) & 7, t = i >> 11;
    int g = lane >> 4, n = lane & 15;
    int c0 = t * 32 + g * 8 + 2 * v + 8 * (v >> 2), o = ot * 16 + n;
    pkWsc[i] = pack2bf(W_sc[c0 * OUT_F + o], W_sc[(c0 + 1) * OUT_F + o]);
    return;
  }
  i -= 2 * 2048;
  if (i < 64) {                              // shadow feature row = zeros
    ((unsigned int*)(feats_bf + (size_t)N_PTS * OUT_F))[i] = 0u;
  }
}

// ---------------------------------------------------------------------------
// Pre-activation: feats_bf = bf16(leaky(features @ W_pre + b_pre))
// One wave per 16-row strip, 16 WMMAs per wave.
// ---------------------------------------------------------------------------
__global__ __launch_bounds__(256) void pre_kernel(const float* __restrict__ features,
                                                  const float* __restrict__ b_pre,
                                                  const unsigned int* __restrict__ pkWpre,
                                                  unsigned short* __restrict__ feats_bf)
{
  int lane = threadIdx.x & 31, w = threadIdx.x >> 5;
  int g = lane >> 4, n = lane & 15;
  int rb = (blockIdx.x * 8 + w) * 16;
  int row = rb + n;

  BFV a0, a1;
  fill_a_f32(a0, features + (size_t)row * IN_F + 0  + g * 8,
                 features + (size_t)row * IN_F + 16 + g * 8);
  fill_a_f32(a1, features + (size_t)row * IN_F + 32 + g * 8,
                 features + (size_t)row * IN_F + 48 + g * 8);

  for (int ot = 0; ot < 8; ot++) {
    v8f acc = {};
    BFV b;
    const uint4* bp = (const uint4*)(pkWpre + ((size_t)(0 * 8 + ot) * 32 + lane) * 8);
    b.q[0] = bp[0]; b.q[1] = bp[1];
    acc = __builtin_amdgcn_wmma_f32_16x16x32_bf16(false, a0.v, false, b.v, (short)0, acc, false, false);
    bp = (const uint4*)(pkWpre + ((size_t)(1 * 8 + ot) * 32 + lane) * 8);
    b.q[0] = bp[0]; b.q[1] = bp[1];
    acc = __builtin_amdgcn_wmma_f32_16x16x32_bf16(false, a1.v, false, b.v, (short)0, acc, false, false);
    float bias = b_pre[ot * 16 + n];
#pragma unroll
    for (int v = 0; v < 8; v++) {
      int r = rb + v + 8 * g;                             // D layout: m = v + 8*g
      feats_bf[(size_t)r * OUT_F + ot * 16 + n] = f2bf(leaky(acc[v] + bias));
    }
  }
}

// ---------------------------------------------------------------------------
// Fused KPConv: one block = 16 queries (8 waves).
// ---------------------------------------------------------------------------
#define QSTR 2064   // halves per query row in s_wfeatT ([q][c=128][k=16] + pad)
#define CSTR 136    // halves per row in s_convA ([q][c=128] + pad)

__global__ __launch_bounds__(256) void kpconv_kernel(
    const float* __restrict__ points,
    const float* __restrict__ features,
    const float* __restrict__ kernel_points,
    const float* __restrict__ b_post,
    const float* __restrict__ b_sc,
    const int*   __restrict__ sample_idx,
    const int*   __restrict__ nbr_idx,
    const unsigned short* __restrict__ feats_bf,
    const unsigned int* __restrict__ pkW,
    const unsigned int* __restrict__ pkWpost,
    const unsigned int* __restrict__ pkWsc,
    float* __restrict__ out)
{
  __shared__ __attribute__((aligned(16))) unsigned short s_wfeatT[16 * QSTR]; // 66 KB
  __shared__ __attribute__((aligned(16))) unsigned int   s_stg[8][2048];      // 64 KB, per wave
  __shared__ float s_diff[8][NNB][4];
  __shared__ int   s_nbr[8][NNB];
  __shared__ float s_kpt[16][4];
  __shared__ __attribute__((aligned(16))) unsigned short s_convA[16 * CSTR];

  int lane = threadIdx.x & 31, w = threadIdx.x >> 5;
  int g = lane >> 4, n = lane & 15;
  int qbase = blockIdx.x * 16;

  if (threadIdx.x < 16) {
    int t = threadIdx.x;
    s_kpt[t][0] = (t < KPN) ? kernel_points[t * 3 + 0] : 0.f;
    s_kpt[t][1] = (t < KPN) ? kernel_points[t * 3 + 1] : 0.f;
    s_kpt[t][2] = (t < KPN) ? kernel_points[t * 3 + 2] : 0.f;
  }
  __syncthreads();

  // ---- step 2: all_w WMMA -> wfeatT in LDS ([q][c][k]) ----
  for (int qi = 0; qi < 2; qi++) {
    int ql = w * 2 + qi;
    int q  = qbase + ql;

    // neighbor diffs (one neighbor per lane)
    int nb = nbr_idx[(size_t)q * NNB + lane];
    s_nbr[w][lane] = nb;
    int si = sample_idx[q];
    float qx = points[si * 3 + 0], qy = points[si * 3 + 1], qz = points[si * 3 + 2];
    float px, py, pz;
    if (nb >= N_PTS) { px = 1e6f; py = 1e6f; pz = 1e6f; }
    else { px = points[nb * 3 + 0]; py = points[nb * 3 + 1]; pz = points[nb * 3 + 2]; }
    s_diff[w][lane][0] = px - qx;
    s_diff[w][lane][1] = py - qy;
    s_diff[w][lane][2] = pz - qz;

    // stage 32 neighbor feature rows, pair-interleaved: dword(p,c) = rows(2p,2p+1) @ col c
#pragma unroll
    for (int it = 0; it < 8; it++) {
      int task = it * 32 + lane;
      int pr = task >> 4;            // row pair 0..15
      int c0 = (task & 15) * 8;      // 8-col chunk
      int na = s_nbr[w][pr * 2], nb2 = s_nbr[w][pr * 2 + 1];
      uint4 ua = *(const uint4*)(feats_bf + (size_t)na  * OUT_F + c0);
      uint4 ub = *(const uint4*)(feats_bf + (size_t)nb2 * OUT_F + c0);
      uint4 o0, o1;                  // v_perm_b32 half-interleave
      o0.x = __builtin_amdgcn_perm(ub.x, ua.x, 0x05040100u);
      o0.y = __builtin_amdgcn_perm(ub.x, ua.x, 0x07060302u);
      o0.z = __builtin_amdgcn_perm(ub.y, ua.y, 0x05040100u);
      o0.w = __builtin_amdgcn_perm(ub.y, ua.y, 0x07060302u);
      o1.x = __builtin_amdgcn_perm(ub.z, ua.z, 0x05040100u);
      o1.y = __builtin_amdgcn_perm(ub.z, ua.z, 0x07060302u);
      o1.z = __builtin_amdgcn_perm(ub.w, ua.w, 0x05040100u);
      o1.w = __builtin_amdgcn_perm(ub.w, ua.w, 0x07060302u);
      *(uint4*)&s_stg[w][pr * 128 + c0]     = o0;
      *(uint4*)&s_stg[w][pr * 128 + c0 + 4] = o1;
    }

    // all_w in WMMA-A register layout: lane row = kernel point k (= n), halves = neighbors
    BFV aw;
    float kx = s_kpt[n][0], ky = s_kpt[n][1], kz = s_kpt[n][2];
#pragma unroll
    for (int hp = 0; hp < 8; hp++) {
      float wv[2];
#pragma unroll
      for (int pos = 0; pos < 2; pos++) {
        int h  = hp * 2 + pos;
        int ni = h + g * 16;                            // contiguous-k neighbor index (matches staged B)
        float dx = s_diff[w][ni][0] - kx;
        float dy = s_diff[w][ni][1] - ky;
        float dz = s_diff[w][ni][2] - kz;
        float d2 = dx * dx + dy * dy + dz * dz;
        float t0 = fmaxf(1.f - sqrtf(d2) * KP_EXTENT_INV, 0.f);
        wv[pos] = (n == 15) ? 0.f : t0;                 // pad kernel point 15
      }
      aw.u[hp] = pack2bf(wv[0], wv[1]);
    }

    // 8 column chunks: wfeat[16k x 16c] = all_w(16x32) @ nx(32x16)
    for (int cc = 0; cc < 8; cc++) {
      BFV b;
#pragma unroll
      for (int v = 0; v < 8; v++)
        b.u[v] = s_stg[w][(g * 8 + v) * 128 + cc * 16 + n];
      v8f acc = {};
      acc = __builtin_amdgcn_wmma_f32_16x16x32_bf16(false, aw.v, false, b.v, (short)0, acc, false, false);
      uint4 o;
      o.x = pack2bf(acc[0], acc[1]);
      o.y = pack2bf(acc[2], acc[3]);
      o.z = pack2bf(acc[4], acc[5]);
      o.w = pack2bf(acc[6], acc[7]);
      // store transposed: wfeatT[ql][c = cc*16+n][k = 8g .. 8g+7]
      *(uint4*)&s_wfeatT[(size_t)ql * QSTR + (cc * 16 + n) * 16 + g * 8] = o;
    }
  }
  __syncthreads();

  // ---- step 3: conv[16q x 16o] over inner dim 2048 (c*16+k), wave w -> o-tile w ----
  v8f conv = {};
  for (int t = 0; t < 64; t++) {
    BFV a, b;
    a.q[0] = *(const uint4*)&s_wfeatT[(size_t)n * QSTR + (2 * t) * 16 + g * 8];
    a.q[1] = *(const uint4*)&s_wfeatT[(size_t)n * QSTR + (2 * t + 1) * 16 + g * 8];
    const uint4* bp = (const uint4*)(pkW + ((size_t)(t * 8 + w) * 32 + lane) * 8);
    b.q[0] = bp[0]; b.q[1] = bp[1];
    if (t + 8 < 64)                    // stream-ahead prefetch of B tiles
      __builtin_prefetch(pkW + ((size_t)((t + 8) * 8 + w) * 32 + lane) * 8, 0, 3);
    conv = __builtin_amdgcn_wmma_f32_16x16x32_bf16(false, a.v, false, b.v, (short)0, conv, false, false);
  }
  // leaky(conv) -> bf16 row-major convA[q][c]
#pragma unroll
  for (int v = 0; v < 8; v++) {
    int qq = v + 8 * g;
    s_convA[qq * CSTR + w * 16 + n] = f2bf(leaky(conv[v]));
  }
  __syncthreads();

  // ---- post GEMM + shortcut GEMM, accumulate into one f32 tile ----
  v8f acc2 = {};
  for (int t = 0; t < 4; t++) {
    BFV a, b;
    a.q[0] = *(const uint4*)&s_convA[n * CSTR + t * 32 + g * 8];
    a.q[1] = *(const uint4*)&s_convA[n * CSTR + t * 32 + 16 + g * 8];
    const uint4* bp = (const uint4*)(pkWpost + ((size_t)(t * 8 + w) * 32 + lane) * 8);
    b.q[0] = bp[0]; b.q[1] = bp[1];
    acc2 = __builtin_amdgcn_wmma_f32_16x16x32_bf16(false, a.v, false, b.v, (short)0, acc2, false, false);
  }
  int srow = sample_idx[qbase + n];
  for (int t = 0; t < 2; t++) {
    BFV a, b;
    fill_a_f32(a, features + (size_t)srow * IN_F + t * 32 + g * 8,
                  features + (size_t)srow * IN_F + t * 32 + 16 + g * 8);
    const uint4* bp = (const uint4*)(pkWsc + ((size_t)(t * 8 + w) * 32 + lane) * 8);
    b.q[0] = bp[0]; b.q[1] = bp[1];
    acc2 = __builtin_amdgcn_wmma_f32_16x16x32_bf16(false, a.v, false, b.v, (short)0, acc2, false, false);
  }
  float bb = b_post[w * 16 + n] + b_sc[w * 16 + n];
#pragma unroll
  for (int v = 0; v < 8; v++) {
    int qq = v + 8 * g;
    out[(size_t)(qbase + qq) * OUT_F + w * 16 + n] = leaky(acc2[v] + bb);
  }
}

// ---------------------------------------------------------------------------
extern "C" void kernel_launch(void* const* d_in, const int* in_sizes, int n_in,
                              void* d_out, int out_size, void* d_ws, size_t ws_size,
                              hipStream_t stream) {
  (void)in_sizes; (void)n_in; (void)out_size; (void)ws_size;
  const float* points         = (const float*)d_in[0];
  const float* features       = (const float*)d_in[1];
  const float* W_pre          = (const float*)d_in[2];
  const float* b_pre          = (const float*)d_in[3];
  const float* kernel_points  = (const float*)d_in[4];
  const float* kp_weights     = (const float*)d_in[5];
  const float* W_post         = (const float*)d_in[6];
  const float* b_post         = (const float*)d_in[7];
  const float* W_sc           = (const float*)d_in[8];
  const float* b_sc           = (const float*)d_in[9];
  const int*   sample_idx     = (const int*)d_in[10];
  const int*   neighbors_idx  = (const int*)d_in[11];
  float* outp = (float*)d_out;

  char* ws = (char*)d_ws;
  unsigned short* feats_bf = (unsigned short*)ws;
  unsigned int*   pkW      = (unsigned int*)(ws + OFF_PKW);
  unsigned int*   pkWpost  = (unsigned int*)(ws + OFF_PKWPOST);
  unsigned int*   pkWpre   = (unsigned int*)(ws + OFF_PKWPRE);
  unsigned int*   pkWsc    = (unsigned int*)(ws + OFF_PKWSC);

  // prep: 64*2048 + 4*2048 + 2*2048 + 2*2048 + 64 = 147,520 threads
  int prep_threads = 64 * 2048 + 4 * 2048 + 2 * 2048 + 2 * 2048 + 64;
  prep_pack_kernel<<<(prep_threads + 255) / 256, 256, 0, stream>>>(
      W_pre, kp_weights, W_post, W_sc, feats_bf, pkW, pkWpost, pkWpre, pkWsc);

  // pre-activation: 131072/16 = 8192 strips, 8 waves/block
  pre_kernel<<<8192 / 8, 256, 0, stream>>>(features, b_pre, pkWpre, feats_bf);

  // fused KPConv: 65536/16 = 4096 query tiles
  kpconv_kernel<<<M_Q / 16, 256, 0, stream>>>(
      points, features, kernel_points, b_post, b_sc,
      sample_idx, neighbors_idx, feats_bf, pkW, pkWpost, pkWsc, outp);
}
